// QuantumLayer_35029753266203
// MI455X (gfx1250) — matrix operand order repaired and checked by reading, not compiled, not run on hardware
//
#include <hip/hip_runtime.h>
#include <hip/hip_fp16.h>

// Quantum 4-qubit expectation layer for MI455X (gfx1250, wave32).
//
// Restructured as: out = signs @ (M @ s0(x))^2 where
//   - s0(x) is the rank-1 encoded state (tensor product of [cos, sin](x_i/2))
//   - M is the batch-independent 16x16 circuit unitary (2 layers of RY+CNOT)
// Batched 16 rows per wave-tile -> one V_WMMA_F32_16X16X32_F16 per tile
// (K=16 used, upper 16 zero-padded). f16 operands / f32 accumulate.
//
// Memory-bound problem: 32 B traffic per batch row (~33.5 MB total),
// ~1.4 us at 23.3 TB/s; the kernel keeps per-row VALU cost to
// 4 fast sincos + ~30 ops and uses only b128 global accesses.

typedef __attribute__((ext_vector_type(16))) _Float16 v16h;
typedef __attribute__((ext_vector_type(8)))  _Float16 v8h;
typedef __attribute__((ext_vector_type(8)))  float    v8f;

#define NQ  4
#define NL  2
#define DIM 16                 // 2^NQ
#define TILES_PER_WAVE 4
#define BLOCK 256
#define WAVES_PER_BLOCK (BLOCK / 32)

// ---------------------------------------------------------------------------
// Kernel 1: build the 16x16 circuit matrix M (f16, row-major) from weights.
// Column j = circuit applied to basis state |j>. One wave, threads 0..15.
// ---------------------------------------------------------------------------
__global__ void build_unitary_kernel(const float* __restrict__ w,
                                     _Float16* __restrict__ Mh) {
    const int j = threadIdx.x;
    if (j >= DIM) return;
    float v[DIM];
#pragma unroll
    for (int i = 0; i < DIM; ++i) v[i] = (i == j) ? 1.0f : 0.0f;

#pragma unroll
    for (int l = 0; l < NL; ++l) {
        // RY(weights[l][q]) on wire q (wire 0 = most-significant bit)
#pragma unroll
        for (int q = 0; q < NQ; ++q) {
            const float th = 0.5f * w[l * NQ + q];
            const float c = cosf(th), s = sinf(th);
            const int bit = 1 << (3 - q);
#pragma unroll
            for (int i = 0; i < DIM; ++i) {
                if ((i & bit) == 0) {
                    const float a0 = v[i], a1 = v[i | bit];
                    v[i]       = c * a0 - s * a1;
                    v[i | bit] = s * a0 + c * a1;
                }
            }
        }
        // CNOT chain: control q, target q+1 (swap target amps where ctrl=1)
#pragma unroll
        for (int q = 0; q < NQ - 1; ++q) {
            const int cb = 1 << (3 - q);
            const int tb = 1 << (2 - q);
#pragma unroll
            for (int i = 0; i < DIM; ++i) {
                if ((i & cb) && !(i & tb)) {
                    const float tmp = v[i];
                    v[i]      = v[i | tb];
                    v[i | tb] = tmp;
                }
            }
        }
    }
#pragma unroll
    for (int i = 0; i < DIM; ++i) Mh[i * DIM + j] = (_Float16)v[i];
}

// ---------------------------------------------------------------------------
// Kernel 2: main batched evaluation. One wave handles TILES_PER_WAVE tiles of
// 16 batch rows; each tile is a single v_wmma_f32_16x16x32_f16.
// ---------------------------------------------------------------------------
__global__ void __launch_bounds__(BLOCK)
qnn_expval_kernel(const float* __restrict__ x,
                  const _Float16* __restrict__ Mh,
                  float* __restrict__ out,
                  int ntiles) {
    // Per-wave 16x16 f16 staging tile ([k][n]) for the LDS transpose of s0.
    __shared__ __align__(32) _Float16 smem[WAVES_PER_BLOCK][DIM * DIM];

    const int tid  = threadIdx.x;
    const int warp = tid >> 5;
    const int lane = tid & 31;
    const int wid  = blockIdx.x * (BLOCK >> 5) + warp;
    const int lo   = lane & 15;   // row m (A) / col n (B,D)
    const int hi   = lane >> 4;   // K-half (A) / row-half (D)

    // --- A fragment: M rows striped per ISA f16 A 16x32 layout.
    // lanes 0-15 hold K=0..7 in VGPR0-3, lanes 16-31 hold K=8..15;
    // VGPR4-7 (K=16..31) are the zero pad.
    v16h a;
    {
        const v8h am = *(const v8h*)(Mh + lo * DIM + hi * 8);
#pragma unroll
        for (int e = 0; e < 8; ++e)  a[e] = am[e];
#pragma unroll
        for (int e = 8; e < 16; ++e) a[e] = (_Float16)0;
    }

#pragma unroll
    for (int t = 0; t < TILES_PER_WAVE; ++t) {
        const int tile = wid * TILES_PER_WAVE + t;   // uniform across wave
        if (tile >= ntiles) break;
        const int base = tile * DIM;

        // --- Lanes 0-15: compute s0 for one batch row, scatter to LDS [k][n].
        if (lane < DIM) {
            const float4 xv = ((const float4*)x)[base + lane];
            float c0, s0, c1, s1, c2, s2, c3, s3;
            __sincosf(0.5f * xv.x, &s0, &c0);
            __sincosf(0.5f * xv.y, &s1, &c1);
            __sincosf(0.5f * xv.z, &s2, &c2);
            __sincosf(0.5f * xv.w, &s3, &c3);
            const float t01[4] = {c0 * c1, c0 * s1, s0 * c1, s0 * s1};
            const float t23[4] = {c2 * c3, c2 * s3, s2 * c3, s2 * s3};
#pragma unroll
            for (int m = 0; m < DIM; ++m)
                smem[warp][m * DIM + lane] =
                    (_Float16)(t01[m >> 2] * t23[m & 3]);
        }
        // DS ops are in-order within a wave; the wait + memory clobber stops
        // the compiler from hoisting the transposed reads above the stores.
        asm volatile("s_wait_dscnt 0x0" ::: "memory");

        // --- B fragment (32x16 f16): lane holds column n=lo, K=8*hi+e in
        // elements 0-7; elements 8-15 are the K=16..31 pad.
        v16h b;
#pragma unroll
        for (int e = 0; e < 16; ++e) b[e] = (_Float16)0;
#pragma unroll
        for (int e = 0; e < 8; ++e) b[e] = smem[warp][(hi * 8 + e) * DIM + lo];

        // --- Y = M x S0  (EXEC all-ones here: no divergence around WMMA)
        v8f cz = {};
        v8f y = __builtin_amdgcn_wmma_f32_16x16x32_f16(
            false, a, false, b, (short)0, cz, false, false);

        // --- out_w[n] = sum_m sign_w(m) * y[m][n]^2.
        // D layout: this lane holds column n=lo, rows m = r + 8*hi.
        // sign bits: w=0 -> bit3(m)=hi (uniform); w=1,2,3 -> bits of r.
        float q[8];
#pragma unroll
        for (int r = 0; r < 8; ++r) q[r] = y[r] * y[r];
        const float s0123 = (q[0] + q[1]) + (q[2] + q[3]);
        const float s4567 = (q[4] + q[5]) + (q[6] + q[7]);
        float p0 = hi ? -(s0123 + s4567) : (s0123 + s4567);
        float p1 = s0123 - s4567;
        float p2 = (q[0] + q[1] + q[4] + q[5]) - (q[2] + q[3] + q[6] + q[7]);
        float p3 = (q[0] + q[2] + q[4] + q[6]) - (q[1] + q[3] + q[5] + q[7]);
        // Combine row-halves: lane L <-> lane L^16.
        p0 += __shfl_xor(p0, 16, 32);
        p1 += __shfl_xor(p1, 16, 32);
        p2 += __shfl_xor(p2, 16, 32);
        p3 += __shfl_xor(p3, 16, 32);

        if (lane < DIM) {
            float4 o = {p0, p1, p2, p3};
            ((float4*)out)[base + lane] = o;   // out[b][0..3], coalesced b128
        }
    }
}

// ---------------------------------------------------------------------------
extern "C" void kernel_launch(void* const* d_in, const int* in_sizes, int n_in,
                              void* d_out, int out_size, void* d_ws, size_t ws_size,
                              hipStream_t stream) {
    const float* x = (const float*)d_in[0];   // [B, 4] f32
    const float* w = (const float*)d_in[1];   // [2, 4] f32
    float* out     = (float*)d_out;           // [B, 4] f32
    _Float16* Mh   = (_Float16*)d_ws;         // 16x16 f16 = 512 B scratch

    const int batch  = in_sizes[0] / NQ;
    const int ntiles = batch / DIM;           // 1,048,576 / 16 = 65,536

    build_unitary_kernel<<<1, 32, 0, stream>>>(w, Mh);

    const int waves  = (ntiles + TILES_PER_WAVE - 1) / TILES_PER_WAVE;
    const int blocks = (waves + WAVES_PER_BLOCK - 1) / WAVES_PER_BLOCK;
    qnn_expval_kernel<<<blocks, BLOCK, 0, stream>>>(x, Mh, out, ntiles);
}
